// RES_MACEModel_31250182045933
// MI455X (gfx1250) — matrix operand order repaired and checked by reading, not compiled, not run on hardware
//
#include <hip/hip_runtime.h>

// ---------------------------------------------------------------------------
// MACE-like equivariant GNN for MI455X (gfx1250, wave32, WMMA).
//
// Dominant cost: per-edge weight generation GEMM (E x 64 x 45056) ->
// v_wmma_f32_16x16x32_bf16 (f32 accumulate), fused with per-edge weight
// application (never materializes the 1.5 GB wp tensor). W2 is staged once
// as column-major bf16 so each lane's WMMA B-fragment is two contiguous
// 32B chunks (global_load_b128 pairs). All CG/path metadata is constexpr and
// kernels are templated on the path -> fully unrolled, accumulators in VGPRs.
//
// Input ordering assumption (setup_inputs() dict insertion order):
//  0:x 1:node_type 2:edge_index 3:ca_idx 4:ptr 5:emb
//  6..9:fc0{w1,b1,w2,b2} 10..13:fc1{...} 14..17:prod0{w1,w2,w3,lin}
//  18..21:prod1{...} 22..25:pred_w1,pred_b1,pred_w2,pred_b2
// ---------------------------------------------------------------------------

typedef __attribute__((ext_vector_type(16))) __bf16 v16bf;
typedef __attribute__((ext_vector_type(8)))  float  v8f;

#define EMBC   64
#define NNODE  1024
#define H0OFF  0
#define H1OFF  (NNODE*EMBC)                 // floats
#define H2OFF  (NNODE*EMBC + NNODE*EMBC*3)  // floats
#define HSETF  (NNODE*EMBC*9)               // floats per h-set

struct CGE { int a, b, c; float v; };

// Real-basis Clebsch-Gordan, sparse, per path (PATHS order).
// l=1 basis (y,z,x); l=2 basis (xy,yz,3z^2-1,xz,x^2-y^2). Unit Frobenius
// norm per c-slice (property of the unitary real<-complex transform).
constexpr CGE KCGT[83] = {
  // P0 (0,0,0)
  {0,0,0, 1.0f},
  // P1 (0,1,1)
  {0,0,0,1.0f},{0,1,1,1.0f},{0,2,2,1.0f},
  // P2 (0,2,2)
  {0,0,0,1.0f},{0,1,1,1.0f},{0,2,2,1.0f},{0,3,3,1.0f},{0,4,4,1.0f},
  // P3 (1,0,1)
  {0,0,0,1.0f},{1,0,1,1.0f},{2,0,2,1.0f},
  // P4 (1,1,0)  -1/sqrt(3)
  {0,0,0,-0.57735027f},{1,1,0,-0.57735027f},{2,2,0,-0.57735027f},
  // P5 (1,1,2)
  {0,0,2,-0.40824829f},{0,0,4,-0.70710678f},
  {1,1,2, 0.81649658f},
  {2,2,2,-0.40824829f},{2,2,4, 0.70710678f},
  {0,2,0, 0.70710678f},{2,0,0, 0.70710678f},
  {0,1,1, 0.70710678f},{1,0,1, 0.70710678f},
  {1,2,3, 0.70710678f},{2,1,3, 0.70710678f},
  // P6 (1,2,1)
  {0,0,2,-0.54772256f},{0,1,1,-0.54772256f},{0,2,0, 0.31622777f},{0,4,0, 0.54772256f},
  {1,1,0,-0.54772256f},{1,2,1,-0.63245553f},{1,3,2,-0.54772256f},
  {2,0,0,-0.54772256f},{2,2,2, 0.31622777f},{2,3,1,-0.54772256f},{2,4,2,-0.54772256f},
  // P7 (2,0,2)
  {0,0,0,1.0f},{1,0,1,1.0f},{2,0,2,1.0f},{3,0,3,1.0f},{4,0,4,1.0f},
  // P8 (2,1,1) = P6 with a<->b swapped
  {0,0,2,-0.54772256f},{1,0,1,-0.54772256f},{2,0,0, 0.31622777f},{4,0,0, 0.54772256f},
  {1,1,0,-0.54772256f},{2,1,1,-0.63245553f},{3,1,2,-0.54772256f},
  {0,2,0,-0.54772256f},{2,2,2, 0.31622777f},{3,2,1,-0.54772256f},{4,2,2,-0.54772256f},
  // P9 (2,2,0)  1/sqrt(5)
  {0,0,0,0.44721360f},{1,1,0,0.44721360f},{2,2,0,0.44721360f},{3,3,0,0.44721360f},{4,4,0,0.44721360f},
  // P10 (2,2,2)  fully symmetric invariant, slices normalized
  {0,0,2,-0.53452248f},{0,2,0,-0.53452248f},{2,0,0,-0.53452248f},
  {4,4,2,-0.53452248f},{4,2,4,-0.53452248f},{2,4,4,-0.53452248f},
  {2,2,2, 0.53452248f},
  {1,1,2, 0.26726124f},{1,2,1, 0.26726124f},{2,1,1, 0.26726124f},
  {3,3,2, 0.26726124f},{3,2,3, 0.26726124f},{2,3,3, 0.26726124f},
  {1,1,4,-0.46291005f},{1,4,1,-0.46291005f},{4,1,1,-0.46291005f},
  {3,3,4, 0.46291005f},{3,4,3, 0.46291005f},{4,3,3, 0.46291005f},
  {0,1,3, 0.46291005f},{0,3,1, 0.46291005f},{1,0,3, 0.46291005f},
  {1,3,0, 0.46291005f},{3,0,1, 0.46291005f},{3,1,0, 0.46291005f},
};
constexpr int KCGOFF[12] = {0,1,4,9,12,15,26,37,42,53,58,83};
constexpr int KPL1[11] = {0,0,0,1,1,1,1,2,2,2,2};
constexpr int KPL2[11] = {0,1,2,0,1,1,2,0,1,2,2};
constexpr int KPL3[11] = {0,1,2,1,0,2,1,2,1,0,2};
constexpr int KSHB[3]  = {0,1,4};   // sh offset per l
constexpr int KOB[3]   = {0,1,4};   // offset of l-block inside a 9-vector
// path id for each of the 83 CG entries (flat iteration in k_prod)
constexpr int KENTP[83] = {
  0, 1,1,1, 2,2,2,2,2, 3,3,3, 4,4,4,
  5,5,5,5,5,5,5,5,5,5,5,
  6,6,6,6,6,6,6,6,6,6,6,
  7,7,7,7,7,
  8,8,8,8,8,8,8,8,8,8,8,
  9,9,9,9,9,
  10,10,10,10,10,10,10,10,10,10,10,10,10,
  10,10,10,10,10,10,10,10,10,10,10,10
};

// --- bf16 helpers (RNE float->bf16) ----------------------------------------
__device__ __forceinline__ unsigned short f2bf(float f) {
  unsigned int u = __float_as_uint(f);
  return (unsigned short)((u + 0x7FFFu + ((u >> 16) & 1u)) >> 16);
}

// ---------------------------------------------------------------------------
// K0: per-edge geometry: spherical harmonics (l=0,1,2) + radial Bessel*cutoff
// ---------------------------------------------------------------------------
__global__ void k_edge(const float* __restrict__ x, const int* __restrict__ src,
                       const int* __restrict__ dst, float* __restrict__ sh,
                       float* __restrict__ ef, int E)
{
  int e = blockIdx.x * blockDim.x + threadIdx.x;
  if (e >= E) return;
  int s = src[e], d = dst[e];
  float dx = x[s*3+0]-x[d*3+0];
  float dy = x[s*3+1]-x[d*3+1];
  float dz = x[s*3+2]-x[d*3+2];
  float r  = sqrtf(dx*dx + dy*dy + dz*dz);
  float inv = 1.0f / r;
  float ux = dx*inv, uy = dy*inv, uz = dz*inv;
  float* S = sh + e*9;
  S[0] = 1.0f;
  S[1] = 1.7320508f*uy;  S[2] = 1.7320508f*uz;  S[3] = 1.7320508f*ux;
  S[4] = 3.8729833f*ux*uy;
  S[5] = 3.8729833f*uy*uz;
  S[6] = 1.1180340f*(3.0f*uz*uz - 1.0f);
  S[7] = 3.8729833f*ux*uz;
  S[8] = 1.9364917f*(ux*ux - uy*uy);
  float uR = r * 0.1f;     // r / R_MAX
  float cut = 0.0f;
  if (uR < 1.0f) {
    float u5 = uR*uR*uR*uR*uR;
    cut = 1.0f - 21.0f*u5 + 35.0f*u5*uR - 15.0f*u5*uR*uR;   // p=5 cutoff
  }
  const float pref = 0.44721360f;  // sqrt(2/R_MAX)
  #pragma unroll
  for (int k = 1; k <= 8; ++k)
    ef[e*8 + k - 1] = pref * sinf((float)k * 3.14159265358979f * r * 0.1f) * inv * cut;
}

// ---------------------------------------------------------------------------
// K1: z = relu(ef @ w1 + b1), stored bf16 (WMMA A-operand feed)
// ---------------------------------------------------------------------------
__global__ void k_z(const float* __restrict__ ef, const float* __restrict__ w1,
                    const float* __restrict__ b1, unsigned short* __restrict__ z, int E)
{
  int i = blockIdx.x * blockDim.x + threadIdx.x;
  if (i >= E * EMBC) return;
  int e = i >> 6, v = i & 63;
  float s = b1[v];
  #pragma unroll
  for (int j = 0; j < 8; ++j) s += ef[e*8 + j] * w1[j*EMBC + v];
  z[i] = f2bf(fmaxf(s, 0.0f));
}

// generic helpers ------------------------------------------------------------
// transpose + convert: d[col*64 + j] = bf16(s[j*numel + col])  (WMMA B layout)
__global__ void k_cvt_t(const float* __restrict__ s, unsigned short* __restrict__ d,
                        int numel) {
  int i = blockIdx.x * blockDim.x + threadIdx.x;   // i = j*numel + col (coalesced read)
  if (i >= numel * EMBC) return;
  int j = i / numel, col = i - j * numel;
  d[(size_t)col * EMBC + j] = f2bf(s[i]);
}
__global__ void k_zero(float* __restrict__ p, int n) {
  int i = blockIdx.x * blockDim.x + threadIdx.x;
  if (i < n) p[i] = 0.0f;
}
__global__ void k_inith0(const float* __restrict__ emb, const int* __restrict__ nt,
                         float* __restrict__ h0, int total) {
  int i = blockIdx.x * blockDim.x + threadIdx.x;
  if (i < total) { int n = i >> 6, u = i & 63; h0[i] = emb[nt[n]*EMBC + u]; }
}

// ---------------------------------------------------------------------------
// K2: fused conv layer, templated on path P (all shapes compile-time).
// Workgroup = 16-edge tile; 4 waves, each owns a 16-wide v-block.
// wp = Z@W2p via v_wmma_f32_16x16x32_bf16, consumed in-register by the
// rank-1 update against the CG-contracted tmp; scatter-add (segment_sum
// over src) via global f32 atomics.
// ---------------------------------------------------------------------------
template<int P>
__global__ __launch_bounds__(128) void k_conv(
    const unsigned short* __restrict__ zbf,   // [E][64] bf16
    const unsigned short* __restrict__ w2t,   // [numel][64] bf16 (col-major)
    const float* __restrict__ b2,             // [numel]
    const float* __restrict__ sh,             // [E][9]
    const float* __restrict__ hcur,           // current h set
    float* __restrict__ hu,                   // conv output h set (atomics)
    const int* __restrict__ src, const int* __restrict__ dst,
    float scale)
{
  constexpr int l1 = KPL1[P], l2 = KPL2[P], l3 = KPL3[P];
  constexpr int n1 = 2*l1+1, n2 = 2*l2+1, n3 = 2*l3+1;
  constexpr int colOff = P * EMBC * EMBC;

  __shared__ unsigned short zt[16*EMBC];
  __shared__ float tmp[16][EMBC][n3];
  __shared__ int dstl[16], srcl[16];

  const int e0  = blockIdx.x * 16;
  const int tid = threadIdx.x;

  if (tid < 16) { dstl[tid] = dst[e0+tid]; srcl[tid] = src[e0+tid]; }
  for (int i = tid; i < 16*EMBC; i += 128)
    zt[i] = zbf[(e0 + (i >> 6))*EMBC + (i & 63)];
  __syncthreads();

  // tmp[e][u][k] = scale * sum_{a,b} h_l1[dst,u,a] * sh_l2[e,b] * CG[a,b,k]
  for (int i = tid; i < 16*EMBC; i += 128) {
    const int el = i >> 6, u = i & 63;
    const int d = dstl[el];
    float hg[n1], shv[n2], t[n3];
    const float* hb = (l1 == 0) ? (hcur + H0OFF + (d*EMBC + u)) :
                      (l1 == 1) ? (hcur + H1OFF + (d*EMBC + u)*3) :
                                  (hcur + H2OFF + (d*EMBC + u)*5);
    #pragma unroll
    for (int q = 0; q < n1; ++q) hg[q] = hb[q];
    const float* sb = sh + (e0+el)*9 + KSHB[l2];
    #pragma unroll
    for (int q = 0; q < n2; ++q) shv[q] = sb[q];
    #pragma unroll
    for (int q = 0; q < n3; ++q) t[q] = 0.0f;
    #pragma unroll
    for (int q = KCGOFF[P]; q < KCGOFF[P+1]; ++q)
      t[KCGT[q].c] += hg[KCGT[q].a] * shv[KCGT[q].b] * KCGT[q].v;
    #pragma unroll
    for (int q = 0; q < n3; ++q) tmp[el][u][q] = t[q] * scale;
  }
  __syncthreads();

  const int lane = tid & 31;
  const int v0   = (tid >> 5) * 16;
  const int m    = lane & 15;
  const bool hi  = lane >= 16;

  // A fragments (16e x 32j bf16), contiguous k = 16*half + s, same order as B
  v16bf a0, a1;
  {
    const int jo = hi ? 16 : 0;
    #pragma unroll
    for (int h = 0; h < 8; ++h) {
      a0[h]   = __builtin_bit_cast(__bf16, zt[m*EMBC +  0 + jo + h]);
      a0[h+8] = __builtin_bit_cast(__bf16, zt[m*EMBC +  8 + jo + h]);
      a1[h]   = __builtin_bit_cast(__bf16, zt[m*EMBC + 32 + jo + h]);
      a1[h+8] = __builtin_bit_cast(__bf16, zt[m*EMBC + 40 + jo + h]);
    }
  }

  float acc[8][n3];
  #pragma unroll
  for (int r = 0; r < 8; ++r)
    #pragma unroll
    for (int k = 0; k < n3; ++k) acc[r][k] = 0.0f;

  const int jb = hi ? 16 : 0;    // K-half owned by this lane group
  for (int u = 0; u < EMBC; ++u) {
    const int col = colOff + u*EMBC + v0 + m;
    // B fragments: contiguous 32B chunks (w2t is col-major) -> b128 loads
    const v16bf* bp = (const v16bf*)(w2t + (size_t)col * EMBC + jb);
    const v16bf b0 = bp[0];            // j = jb .. jb+15
    const v16bf b1 = bp[2];            // j = 32+jb .. 32+jb+15
    __builtin_prefetch(w2t + (size_t)(col + EMBC) * EMBC + jb, 0, 1);

    v8f c = {0.f,0.f,0.f,0.f,0.f,0.f,0.f,0.f};
    c = __builtin_amdgcn_wmma_f32_16x16x32_bf16(false, a0, false, b0, (short)0, c, false, false);
    c = __builtin_amdgcn_wmma_f32_16x16x32_bf16(false, a1, false, b1, (short)0, c, false, false);

    const float b2v = b2[col];
    #pragma unroll
    for (int r = 0; r < 8; ++r) {
      const float wp = c[r] + b2v;
      const int er = r + (hi ? 8 : 0);
      #pragma unroll
      for (int k = 0; k < n3; ++k) acc[r][k] += wp * tmp[er][u][k];
    }
  }

  float* hb = (l3 == 0) ? (hu + H0OFF) : (l3 == 1) ? (hu + H1OFF) : (hu + H2OFF);
  const int v = v0 + m;
  #pragma unroll
  for (int r = 0; r < 8; ++r) {
    const int er  = r + (hi ? 8 : 0);
    const int row = srcl[er];
    #pragma unroll
    for (int k = 0; k < n3; ++k)
      atomicAdd(hb + (row*EMBC + v)*n3 + k, acc[r][k]);
  }
}

// ---------------------------------------------------------------------------
// K3: product block. Fully unrolled flat CG iteration -> straight-line FMAs.
// X,Y,O are 9-vectors: [l0 | l1(3) | l2(5)].
// ---------------------------------------------------------------------------
__device__ __forceinline__ void tpacc9(const float (&X)[9], const float (&Y)[9],
                                       const float* __restrict__ wt, int u,
                                       float (&O)[9])
{
  #pragma unroll
  for (int q = 0; q < 83; ++q) {
    const int p = KENTP[q];
    O[KOB[KPL3[p]] + KCGT[q].c] +=
        wt[p*EMBC + u] * X[KOB[KPL1[p]] + KCGT[q].a] * Y[KOB[KPL2[p]] + KCGT[q].b] * KCGT[q].v;
  }
}

__global__ __launch_bounds__(64) void k_prod(
    const float* __restrict__ A, const float* __restrict__ sc, float* __restrict__ hn,
    const float* __restrict__ w1, const float* __restrict__ w2,
    const float* __restrict__ w3, const float* __restrict__ lin)
{
  const int n = blockIdx.x, u = threadIdx.x;
  float Av[9], B2v[9], B3v[9];
  Av[0] = A[H0OFF + n*EMBC + u];
  #pragma unroll
  for (int i = 0; i < 3; ++i) Av[1+i] = A[H1OFF + (n*EMBC + u)*3 + i];
  #pragma unroll
  for (int i = 0; i < 5; ++i) Av[4+i] = A[H2OFF + (n*EMBC + u)*5 + i];
  #pragma unroll
  for (int i = 0; i < 9; ++i) { B2v[i] = 0.0f; B3v[i] = 0.0f; }

  tpacc9(Av, Av,  w2, u, B2v);
  tpacc9(Av, B2v, w3, u, B3v);

  __shared__ float S[EMBC][9];
  S[u][0] = w1[0*EMBC + u]*Av[0] + B2v[0] + B3v[0];
  #pragma unroll
  for (int i = 0; i < 3; ++i) S[u][1+i] = w1[1*EMBC + u]*Av[1+i] + B2v[1+i] + B3v[1+i];
  #pragma unroll
  for (int i = 0; i < 5; ++i) S[u][4+i] = w1[2*EMBC + u]*Av[4+i] + B2v[4+i] + B3v[4+i];
  __syncthreads();

  const int v = u;
  float r[9] = {0,0,0,0,0,0,0,0,0};
  for (int uu = 0; uu < EMBC; ++uu) {
    const float l0 = lin[0*EMBC*EMBC + uu*EMBC + v];
    const float l1 = lin[1*EMBC*EMBC + uu*EMBC + v];
    const float l2 = lin[2*EMBC*EMBC + uu*EMBC + v];
    r[0] += S[uu][0]*l0;
    #pragma unroll
    for (int i = 0; i < 3; ++i) r[1+i] += S[uu][1+i]*l1;
    #pragma unroll
    for (int i = 0; i < 5; ++i) r[4+i] += S[uu][4+i]*l2;
  }
  hn[H0OFF + n*EMBC + v] = r[0]*0.125f + sc[H0OFF + n*EMBC + v];
  #pragma unroll
  for (int i = 0; i < 3; ++i)
    hn[H1OFF + (n*EMBC + v)*3 + i] = r[1+i]*0.125f + sc[H1OFF + (n*EMBC + v)*3 + i];
  #pragma unroll
  for (int i = 0; i < 5; ++i)
    hn[H2OFF + (n*EMBC + v)*5 + i] = r[4+i]*0.125f + sc[H2OFF + (n*EMBC + v)*5 + i];
}

// ---------------------------------------------------------------------------
// K4: readout head for the 4 selected nodes only
// ---------------------------------------------------------------------------
__global__ __launch_bounds__(64) void k_pred(
    const float* __restrict__ h0, const int* __restrict__ ca, const int* __restrict__ ptr,
    const float* __restrict__ W1, const float* __restrict__ B1,
    const float* __restrict__ W2, const float* __restrict__ B2, float* __restrict__ out)
{
  const int g = blockIdx.x, v = threadIdx.x;
  const int n = ca[g] + ptr[g];
  __shared__ float t1[EMBC];
  float s = B1[v];
  for (int u = 0; u < EMBC; ++u) s += h0[n*EMBC + u] * W1[u*EMBC + v];
  t1[v] = fmaxf(s, 0.0f);
  __syncthreads();
  if (v < 20) {
    float o = B2[v];
    for (int j = 0; j < EMBC; ++j) o += t1[j] * W2[j*20 + v];
    out[g*20 + v] = o;
  }
}

// ---------------------------------------------------------------------------
extern "C" void kernel_launch(void* const* d_in, const int* in_sizes, int n_in,
                              void* d_out, int out_size, void* d_ws, size_t ws_size,
                              hipStream_t stream)
{
  (void)n_in; (void)out_size; (void)ws_size;
  const float* x    = (const float*)d_in[0];
  const int*   ntyp = (const int*)  d_in[1];
  const int*   eidx = (const int*)  d_in[2];
  const int*   ca   = (const int*)  d_in[3];
  const int*   ptr  = (const int*)  d_in[4];
  const float* emb  = (const float*)d_in[5];
  const float* f0w1 = (const float*)d_in[6];
  const float* f0b1 = (const float*)d_in[7];
  const float* f0w2 = (const float*)d_in[8];
  const float* f0b2 = (const float*)d_in[9];
  const float* f1w1 = (const float*)d_in[10];
  const float* f1b1 = (const float*)d_in[11];
  const float* f1w2 = (const float*)d_in[12];
  const float* f1b2 = (const float*)d_in[13];
  const float* p0w1 = (const float*)d_in[14];
  const float* p0w2 = (const float*)d_in[15];
  const float* p0w3 = (const float*)d_in[16];
  const float* p0li = (const float*)d_in[17];
  const float* p1w1 = (const float*)d_in[18];
  const float* p1w2 = (const float*)d_in[19];
  const float* p1w3 = (const float*)d_in[20];
  const float* p1li = (const float*)d_in[21];
  const float* pW1  = (const float*)d_in[22];
  const float* pB1  = (const float*)d_in[23];
  const float* pW2  = (const float*)d_in[24];
  const float* pB2  = (const float*)d_in[25];

  const int E = in_sizes[2] / 2;   // 8192
  const int N = in_sizes[1];       // 1024
  const int G = in_sizes[3];       // 4
  const int* src = eidx;
  const int* dst = eidx + E;

  char* ws = (char*)d_ws;
  float*          sh   = (float*)(ws + 0);                    // E*9*4  = 294912
  float*          ef   = (float*)(ws + 294912);               // E*8*4  = 262144
  unsigned short* zbf  = (unsigned short*)(ws + 557056);      // E*64*2 = 1048576
  unsigned short* w20t = (unsigned short*)(ws + 1605632);     // 786432*2
  unsigned short* w21t = (unsigned short*)(ws + 3178496);     // 2883584*2
  float*          hb0  = (float*)(ws + 8945664);              // 3 h-sets
  float*          hb1  = hb0 + HSETF;                         // hu (conv output)
  float*          hb2  = hb1 + HSETF;

  const int n20 = EMBC * 3  * EMBC * EMBC;   // 786432
  const int n21 = EMBC * 11 * EMBC * EMBC;   // 2883584

  // edge geometry + bf16 (col-major) weight staging
  k_edge <<<(E + 255)/256, 256, 0, stream>>>(x, src, dst, sh, ef, E);
  k_cvt_t<<<(n20 + 255)/256, 256, 0, stream>>>(f0w2, w20t, 3*EMBC*EMBC);
  k_cvt_t<<<(n21 + 255)/256, 256, 0, stream>>>(f1w2, w21t, 11*EMBC*EMBC);

  // h init: zero set, then h0 = emb[node_type]
  k_zero  <<<(HSETF + 255)/256, 256, 0, stream>>>(hb0, HSETF);
  k_inith0<<<(N*EMBC + 255)/256, 256, 0, stream>>>(emb, ntyp, hb0 + H0OFF, N*EMBC);

  const dim3 cg(E/16);

  // ---- layer 0: in_ls = (0,), paths 0..2, n_to = {1,1,1} -> scale 1/8 -----
  k_z   <<<(E*EMBC + 255)/256, 256, 0, stream>>>(ef, f0w1, f0b1, zbf, E);
  k_zero<<<(HSETF + 255)/256, 256, 0, stream>>>(hb1, HSETF);
  k_conv<0><<<cg, 128, 0, stream>>>(zbf, w20t, f0b2, sh, hb0, hb1, src, dst, 0.125f);
  k_conv<1><<<cg, 128, 0, stream>>>(zbf, w20t, f0b2, sh, hb0, hb1, src, dst, 0.125f);
  k_conv<2><<<cg, 128, 0, stream>>>(zbf, w20t, f0b2, sh, hb0, hb1, src, dst, 0.125f);
  k_prod<<<N, 64, 0, stream>>>(hb1, hb0, hb2, p0w1, p0w2, p0w3, p0li);

  // ---- layer 1: all 11 paths, n_to = {3,4,4} ------------------------------
  const float s0 = 0.07216878f;   // 1/sqrt(64*3)
  const float s12 = 0.0625f;      // 1/sqrt(64*4)
  k_z   <<<(E*EMBC + 255)/256, 256, 0, stream>>>(ef, f1w1, f1b1, zbf, E);
  k_zero<<<(HSETF + 255)/256, 256, 0, stream>>>(hb1, HSETF);
  k_conv<0> <<<cg, 128, 0, stream>>>(zbf, w21t, f1b2, sh, hb2, hb1, src, dst, s0);
  k_conv<1> <<<cg, 128, 0, stream>>>(zbf, w21t, f1b2, sh, hb2, hb1, src, dst, s12);
  k_conv<2> <<<cg, 128, 0, stream>>>(zbf, w21t, f1b2, sh, hb2, hb1, src, dst, s12);
  k_conv<3> <<<cg, 128, 0, stream>>>(zbf, w21t, f1b2, sh, hb2, hb1, src, dst, s12);
  k_conv<4> <<<cg, 128, 0, stream>>>(zbf, w21t, f1b2, sh, hb2, hb1, src, dst, s0);
  k_conv<5> <<<cg, 128, 0, stream>>>(zbf, w21t, f1b2, sh, hb2, hb1, src, dst, s12);
  k_conv<6> <<<cg, 128, 0, stream>>>(zbf, w21t, f1b2, sh, hb2, hb1, src, dst, s12);
  k_conv<7> <<<cg, 128, 0, stream>>>(zbf, w21t, f1b2, sh, hb2, hb1, src, dst, s12);
  k_conv<8> <<<cg, 128, 0, stream>>>(zbf, w21t, f1b2, sh, hb2, hb1, src, dst, s12);
  k_conv<9> <<<cg, 128, 0, stream>>>(zbf, w21t, f1b2, sh, hb2, hb1, src, dst, s0);
  k_conv<10><<<cg, 128, 0, stream>>>(zbf, w21t, f1b2, sh, hb2, hb1, src, dst, s12);
  k_prod<<<N, 64, 0, stream>>>(hb1, hb2, hb0, p1w1, p1w2, p1w3, p1li);

  // ---- readout ------------------------------------------------------------
  k_pred<<<G, 64, 0, stream>>>(hb0 + H0OFF, ca, ptr, pW1, pB1, pW2, pB2, (float*)d_out);
}
